// HybridQLSTM_65481071408496
// MI455X (gfx1250) — matrix operand order repaired and checked by reading, not compiled, not run on hardware
//
#include <hip/hip_runtime.h>
#include <math.h>

// ---------------------------------------------------------------------------
// Hybrid QLSTM for MI455X (gfx1250, wave32).
//   K1: 401,408 independent 4-qubit quanvolution circuits (complex state),
//       writing feats as (2048 x 800) with zero-padded K tail (784..799).
//   K2: pre_x GEMM (2048x800) @ (800x16) via v_wmma_f32_16x16x32_f16 —
//       guard-free inner loop (both operands zero-padded to K=800).
//   K3: single-workgroup sequential LSTM scan (64 steps) with LDS state,
//       128 threads = 4 gates x 32 batch, real-amplitude qubit sim + classifier.
// ---------------------------------------------------------------------------

typedef __attribute__((ext_vector_type(16))) _Float16 v16h;
typedef __attribute__((ext_vector_type(8)))  float    v8f;

#define B_SZ    32
#define S_LEN   64
#define N_IMGS  (B_SZ * S_LEN)      // 2048
#define FEAT_D  784
#define KPAD    800                  // 784 padded to multiple of 32
#define D_IN    788

// ------------------------- complex 4-qubit helpers -------------------------
__device__ __forceinline__ void ry_c(float2* st, int w, float t) {
    float c = cosf(0.5f * t), s = sinf(0.5f * t);
    int m = 1 << (3 - w);
#pragma unroll
    for (int i = 0; i < 16; ++i) {
        if (!(i & m)) {
            int j = i | m;
            float2 a0 = st[i], a1 = st[j];
            st[i] = make_float2(c * a0.x - s * a1.x, c * a0.y - s * a1.y);
            st[j] = make_float2(s * a0.x + c * a1.x, s * a0.y + c * a1.y);
        }
    }
}

__device__ __forceinline__ void rx_c(float2* st, int w, float t) {
    float c = cosf(0.5f * t), s = sinf(0.5f * t);
    int m = 1 << (3 - w);
#pragma unroll
    for (int i = 0; i < 16; ++i) {
        if (!(i & m)) {
            int j = i | m;
            float2 a0 = st[i], a1 = st[j];
            // new0 = c*a0 + (-i s)*a1 ; new1 = (-i s)*a0 + c*a1
            st[i] = make_float2(c * a0.x + s * a1.y, c * a0.y - s * a1.x);
            st[j] = make_float2(c * a1.x + s * a0.y, c * a1.y - s * a0.x);
        }
    }
}

__device__ __forceinline__ void rz_c(float2* st, int w, float t) {
    float ce = cosf(0.5f * t), se = sinf(0.5f * t);
    int m = 1 << (3 - w);
#pragma unroll
    for (int i = 0; i < 16; ++i) {
        if (!(i & m)) {
            int j = i | m;
            float2 a0 = st[i], a1 = st[j];
            st[i] = make_float2(ce * a0.x + se * a0.y, ce * a0.y - se * a0.x); // e^{-it/2}*a0
            st[j] = make_float2(ce * a1.x - se * a1.y, ce * a1.y + se * a1.x); // e^{+it/2}*a1
        }
    }
}

__device__ __forceinline__ void cnot_c(float2* st, int c, int tg) {
    int mc = 1 << (3 - c), mt = 1 << (3 - tg);
#pragma unroll
    for (int i = 0; i < 16; ++i) {
        if ((i & mc) && !(i & mt)) {
            int j = i | mt;
            float2 tmp = st[i]; st[i] = st[j]; st[j] = tmp;
        }
    }
}

// ------------------ Kernel 1: quanvolution (one thread/patch) --------------
__global__ __launch_bounds__(256)
void quanv_kernel(const float* __restrict__ x,
                  const float* __restrict__ kp,
                  float* __restrict__ feats)   // (2048, KPAD)
{
    int tid = blockIdx.x * blockDim.x + threadIdx.x;     // 0 .. 401407
    if (tid >= N_IMGS * 196) return;
    int n  = tid / 196;
    int p  = tid - n * 196;
    int pr = p / 14, pc = p - pr * 14;
    const float* img = x + (size_t)n * 784;

    // zero the K padding (columns 784..799) once per row, deterministic.
    if (p < KPAD - FEAT_D) feats[(size_t)n * KPAD + FEAT_D + p] = 0.f;

    float ang[4];
    ang[0] = img[(2 * pr)     * 28 + 2 * pc];
    ang[1] = img[(2 * pr)     * 28 + 2 * pc + 1];
    ang[2] = img[(2 * pr + 1) * 28 + 2 * pc];
    ang[3] = img[(2 * pr + 1) * 28 + 2 * pc + 1];

    float k0 = kp[0], k1 = kp[1], k2 = kp[2], k3 = kp[3], k4 = kp[4], k5 = kp[5];

    float2 st[16];
#pragma unroll
    for (int i = 0; i < 16; ++i) st[i] = make_float2(0.f, 0.f);
    st[0] = make_float2(1.f, 0.f);

    // RY encoding
    ry_c(st, 0, ang[0]); ry_c(st, 1, ang[1]); ry_c(st, 2, ang[2]); ry_c(st, 3, ang[3]);
    // fixed random layer
    ry_c(st, 0, k0); rx_c(st, 1, k1); rz_c(st, 2, k2); ry_c(st, 3, k3);
    cnot_c(st, 0, 1); cnot_c(st, 2, 3);
    rx_c(st, 2, k4); ry_c(st, 0, k5);

    // measure <Z_w>
#pragma unroll
    for (int w = 0; w < 4; ++w) {
        int m = 1 << (3 - w);
        float z = 0.f;
#pragma unroll
        for (int i = 0; i < 16; ++i) {
            float pr2 = st[i].x * st[i].x + st[i].y * st[i].y;
            z += (i & m) ? -pr2 : pr2;
        }
        feats[(size_t)n * KPAD + p * 4 + w] = z;
    }
}

// ---- Kernel 2: pre_x = feats(2048xKPAD) @ W16^T(KPADx16) + bias, f16 WMMA --
// One 16x16 output tile per wave; 8 waves/block, 16 blocks -> 128 tiles.
// Guard-free: both A (feats) and B (W in LDS) are zero-padded to K=KPAD.
__global__ __launch_bounds__(256)
void prex_gemm_kernel(const float* __restrict__ feats,  // (2048, KPAD)
                      const float* __restrict__ Wg,     // (4,4,788) row n=g*4+q
                      const float* __restrict__ bg,     // (4,4) flat
                      float* __restrict__ prex)         // (2048,16)
{
    __shared__ _Float16 Bl[16 * KPAD];                  // 25.6 KB, f16 W, K-padded
    for (int e = threadIdx.x; e < 16 * KPAD; e += 256) {
        int n = e / KPAD, k = e - n * KPAD;
        float v = (k < FEAT_D) ? Wg[n * D_IN + k] : 0.f;
        Bl[e] = (_Float16)v;
    }
    __syncthreads();

    int wave = threadIdx.x >> 5;
    int lane = threadIdx.x & 31;
    int tile = blockIdx.x * 8 + wave;                   // 0..127
    int m0   = tile * 16;
    int hi   = lane >> 4;                               // lane group (0/1)
    int mrow = m0 + (lane & 15);
    int nCol = lane & 15;
    const float* rowA = feats + (size_t)mrow * KPAD;

    v8f acc = {0.f, 0.f, 0.f, 0.f, 0.f, 0.f, 0.f, 0.f};

    for (int kc = 0; kc < KPAD; kc += 32) {
        // A fragment: lane holds M row mrow; halves 0..15 <- K = k0..k0+15
        // (k0 = kc + 16*hi).  16B-aligned loads.
        int k0 = kc + hi * 16;
        float4 f0 = *(const float4*)(rowA + k0);
        float4 f1 = *(const float4*)(rowA + k0 + 4);
        float4 f2 = *(const float4*)(rowA + k0 + 8);
        float4 f3 = *(const float4*)(rowA + k0 + 12);

        v16h a;
        a[0]  = (_Float16)f0.x; a[1]  = (_Float16)f0.y;
        a[2]  = (_Float16)f0.z; a[3]  = (_Float16)f0.w;
        a[4]  = (_Float16)f1.x; a[5]  = (_Float16)f1.y;
        a[6]  = (_Float16)f1.z; a[7]  = (_Float16)f1.w;
        a[8]  = (_Float16)f2.x; a[9]  = (_Float16)f2.y;
        a[10] = (_Float16)f2.z; a[11] = (_Float16)f2.w;
        a[12] = (_Float16)f3.x; a[13] = (_Float16)f3.y;
        a[14] = (_Float16)f3.z; a[15] = (_Float16)f3.w;

        // B fragment: lane holds column nCol; 16 contiguous K starting kc+16*hi.
        int kb = kc + hi * 16;
        v16h b = *(const v16h*)&Bl[nCol * KPAD + kb];

        acc = __builtin_amdgcn_wmma_f32_16x16x32_f16(
                  false, a, false, b, (short)0, acc, false, false);
    }

    float bias = bg[nCol];
#pragma unroll
    for (int r = 0; r < 8; ++r) {
        int m = m0 + r + hi * 8;                        // D layout: VGPR r -> M=r(+8)
        prex[(size_t)m * 16 + nCol] = acc[r] + bias;
    }
}

// -------- Kernel 3: sequential LSTM scan, one workgroup of 128 threads -----
__device__ __forceinline__ float sigmoidf(float v) { return 1.f / (1.f + expf(-v)); }

__global__ __launch_bounds__(128)
void qlstm_scan_kernel(const float* __restrict__ prex,  // (2048,16), bias included
                       const float* __restrict__ Wg,    // (4,4,788) -> h part [784..787]
                       const float* __restrict__ gp,    // (4,4) trainable RY params
                       const float* __restrict__ Wc,    // (10,4)
                       const float* __restrict__ bc,    // (10,)
                       float* __restrict__ out)         // (B,S,10)
{
    __shared__ float hs[B_SZ][4], cs[B_SZ][4];
    __shared__ float zl[4][B_SZ][4];
    __shared__ float Whl[16][4];
    __shared__ float gpl[16], Wcl[40], bcl[10];

    int t = threadIdx.x;
    if (t < 64) Whl[t >> 2][t & 3] = Wg[(t >> 2) * D_IN + FEAT_D + (t & 3)];
    if (t < 16) gpl[t] = gp[t];
    if (t < 40) Wcl[t] = Wc[t];
    if (t < 10) bcl[t] = bc[t];
    if (t < B_SZ) {
#pragma unroll
        for (int q = 0; q < 4; ++q) { hs[t][q] = 0.f; cs[t][q] = 0.f; }
    }
    __syncthreads();

    int g = t >> 5;          // gate 0..3 (f,i,update,o)
    int b = t & 31;          // batch lane

    for (int s = 0; s < S_LEN; ++s) {
        int row = b * S_LEN + s;

        // qlayer circuit: RY(angle)+RY(param) fuse -> single RY; state stays real.
        float st[16];
#pragma unroll
        for (int i = 0; i < 16; ++i) st[i] = 0.f;
        st[0] = 1.f;

#pragma unroll
        for (int q = 0; q < 4; ++q) {
            float ang = prex[(size_t)row * 16 + g * 4 + q];
#pragma unroll
            for (int j = 0; j < 4; ++j) ang += Whl[g * 4 + q][j] * hs[b][j];
            float th = 0.5f * (ang + gpl[g * 4 + q]);
            float c = cosf(th), sn = sinf(th);
            int m = 1 << (3 - q);
#pragma unroll
            for (int i = 0; i < 16; ++i) {
                if (!(i & m)) {
                    int j2 = i | m;
                    float a0 = st[i], a1 = st[j2];
                    st[i]  = c * a0 - sn * a1;
                    st[j2] = sn * a0 + c * a1;
                }
            }
        }
        // CNOT ring: (0,1),(1,2),(2,3),(3,0)
#pragma unroll
        for (int w = 0; w < 4; ++w) {
            int ctl = w, tgt = (w + 1) & 3;
            int mc = 1 << (3 - ctl), mt = 1 << (3 - tgt);
#pragma unroll
            for (int i = 0; i < 16; ++i) {
                if ((i & mc) && !(i & mt)) {
                    int j2 = i | mt;
                    float tmp = st[i]; st[i] = st[j2]; st[j2] = tmp;
                }
            }
        }
        // <Z_q>
#pragma unroll
        for (int q = 0; q < 4; ++q) {
            int m = 1 << (3 - q);
            float z = 0.f;
#pragma unroll
            for (int i = 0; i < 16; ++i) {
                float pr2 = st[i] * st[i];
                z += (i & m) ? -pr2 : pr2;
            }
            zl[g][b][q] = z;
        }
        __syncthreads();

        if (t < B_SZ) {
            int bb = t;
            float hnew[4];
#pragma unroll
            for (int q = 0; q < 4; ++q) {
                float f  = sigmoidf(zl[0][bb][q]);
                float ii = sigmoidf(zl[1][bb][q]);
                float gg = tanhf(zl[2][bb][q]);
                float o  = sigmoidf(zl[3][bb][q]);
                float cc = f * cs[bb][q] + ii * gg;
                cs[bb][q] = cc;
                hnew[q] = o * tanhf(cc);
                hs[bb][q] = hnew[q];
            }
            // classifier + log_softmax for (bb, s)
            float lg[10];
            float mx = -1e30f;
#pragma unroll
            for (int k = 0; k < 10; ++k) {
                float v = bcl[k];
#pragma unroll
                for (int q = 0; q < 4; ++q) v += Wcl[k * 4 + q] * hnew[q];
                lg[k] = v;
                mx = fmaxf(mx, v);
            }
            float sum = 0.f;
#pragma unroll
            for (int k = 0; k < 10; ++k) sum += expf(lg[k] - mx);
            float lse = logf(sum);
            float* orow = out + ((size_t)bb * S_LEN + s) * 10;
#pragma unroll
            for (int k = 0; k < 10; ++k) orow[k] = lg[k] - mx - lse;
        }
        __syncthreads();    // hs/cs visible before next step's reads
    }
}

// ------------------------------- launcher ----------------------------------
extern "C" void kernel_launch(void* const* d_in, const int* in_sizes, int n_in,
                              void* d_out, int out_size, void* d_ws, size_t ws_size,
                              hipStream_t stream) {
    (void)in_sizes; (void)n_in; (void)out_size; (void)ws_size;
    const float* x   = (const float*)d_in[0];   // (32,64,1,28,28)
    const float* kp  = (const float*)d_in[1];   // (6,)
    const float* gp  = (const float*)d_in[2];   // (4,4)
    const float* Wg  = (const float*)d_in[3];   // (4,4,788)
    const float* bg  = (const float*)d_in[4];   // (4,4)
    const float* Wc  = (const float*)d_in[5];   // (10,4)
    const float* bc  = (const float*)d_in[6];   // (10,)
    float* out = (float*)d_out;                 // (32,64,10)

    // workspace: feats (2048 x KPAD f32, K-padded) then prex (2048x16 f32)
    float* feats = (float*)d_ws;
    float* prex  = (float*)((char*)d_ws + (size_t)N_IMGS * KPAD * sizeof(float));

    quanv_kernel<<<(N_IMGS * 196) / 256, 256, 0, stream>>>(x, kp, feats);
    prex_gemm_kernel<<<16, 256, 0, stream>>>(feats, Wg, bg, prex);
    qlstm_scan_kernel<<<1, 128, 0, stream>>>(prex, Wg, gp, Wc, bc, out);
}
